// SeqDecoderWithAttn_120259084288
// MI455X (gfx1250) — compile-verified, matrix operand order checked by reading, not run on hardware
//
#include <hip/hip_runtime.h>
#include <hip/hip_bf16.h>

// ---------------------------------------------------------------------------
// SeqDecoderWithAttn for MI455X (gfx1250, wave32, WMMA).
// B=16 T=128 S=256 H=M=512 V=32000 L=2
// - GEMMs: v_wmma_f32_16x16x32_bf16, 128x128 block tile, A K-chunks double-
//   buffered in LDS via global_load_async_to_lds_b128 (ASYNCcnt).
// - GRU h0 init staged by the Tensor Data Mover (tensor_load_to_lds,
//   TENSORcnt) with a hand-packed D# descriptor (ISA 08_async_tensor 8.3/8.4).
// ---------------------------------------------------------------------------

typedef __attribute__((ext_vector_type(16))) __bf16 v16bf;
typedef __attribute__((ext_vector_type(8)))  __bf16 v8bf;
typedef __attribute__((ext_vector_type(8)))  float  v8f;
typedef __attribute__((ext_vector_type(4)))  unsigned int u32x4;
typedef __attribute__((ext_vector_type(8)))  unsigned int u32x8;

#define CB   16      // batch
#define CT   128     // time
#define CS   256     // enc seq
#define CH   512     // embed / hidden width
#define CV   32000   // vocab
#define CBT  (CB*CT) // 2048
#define C3M  (3*CH)  // 1536

#define GMB  128     // GEMM block tile M
#define GNB  128     // GEMM block tile N
#define GKC  64      // GEMM K chunk staged in LDS

__device__ __forceinline__ unsigned short f2bf(float f) {
  unsigned int u = __float_as_uint(f);
  u += 0x7fffu + ((u >> 16) & 1u);          // round-to-nearest-even
  return (unsigned short)(u >> 16);
}

// 16x32 bf16 fragment slice for this lane (ISA 7.12.2): two contiguous 16B
// chunks at +0 and +32 bytes; the lane's +0/+8 K-half is baked into `base`.
__device__ __forceinline__ v16bf load_frag(const unsigned short* base) {
  v8bf lo = *reinterpret_cast<const v8bf*>(base);
  v8bf hi = *reinterpret_cast<const v8bf*>(base + 16);
  return __builtin_shufflevector(lo, hi, 0,1,2,3,4,5,6,7,8,9,10,11,12,13,14,15);
}

__device__ __forceinline__ v8f wmma_bf16(v16bf a, v16bf b, v8f c) {
  return __builtin_amdgcn_wmma_f32_16x16x32_bf16(
      /*neg_a=*/false, a, /*neg_b=*/false, b,
      /*c_mod=*/(short)0, c, /*reuse_a=*/false, /*reuse_b=*/false);
}

// Async DMA: global -> LDS, 16 bytes per lane, tracked by ASYNCcnt.
__device__ __forceinline__ void async_g2l_b128(unsigned lds_off, const void* g) {
  asm volatile("global_load_async_to_lds_b128 %0, %1, off"
               :: "v"(lds_off), "v"(g) : "memory");
}
__device__ __forceinline__ void wait_asynccnt0() {
  asm volatile("s_wait_asynccnt 0x0" ::: "memory");
}

// Tensor Data Mover: 1 contiguous tile of `dwords` f32 from global -> LDS.
// D# packing per ISA 08_async_tensor 8.3 (group0) / 8.4 (group1); 2D-or-less
// tensors use the two-group form (VADDR2/3 NULL).
__device__ __forceinline__ void tdm_load_1d_f32(unsigned lds_byte_addr,
                                                const void* gptr,
                                                unsigned dwords) {
  unsigned long long ga = (unsigned long long)(uintptr_t)gptr;
  u32x4 g0;
  g0[0] = 1u;                                        // count=1, user mode
  g0[1] = lds_byte_addr;                             // lds_addr
  g0[2] = (unsigned)ga;                              // global_addr[31:0]
  g0[3] = (unsigned)((ga >> 32) & 0x01FFFFFFu)       // global_addr[56:32]
          | (2u << 30);                              // type=2 ("image")
  u32x8 g1;
  g1[0] = (2u << 16);                                // data_size=2 (4 bytes)
  g1[1] = (dwords & 0xFFFFu) << 16;                  // tensor_dim0[15:0]
  g1[2] = (dwords >> 16) | (1u << 16);               // tensor_dim0[31:16] | tensor_dim1=1
  g1[3] = (dwords & 0xFFFFu) << 16;                  // tile_dim0 (=dwords, <=65535)
  g1[4] = 1u;                                        // tile_dim1=1, tile_dim2=0
  g1[5] = dwords;                                    // tensor_dim0_stride[31:0]
  g1[6] = 0u;                                        // stride hi | dim1_stride lo
  g1[7] = 0u;
  asm volatile("tensor_load_to_lds %0, %1" :: "s"(g0), "s"(g1) : "memory");
}

// ---------------------------------------------------------------------------
// f32 -> bf16 conversion
// ---------------------------------------------------------------------------
__global__ void cvt_bf16_kernel(const float* __restrict__ src,
                                unsigned short* __restrict__ dst, int n) {
  int i = blockIdx.x * blockDim.x + threadIdx.x;
  if (i < n) dst[i] = f2bf(src[i]);
}

// ---------------------------------------------------------------------------
// Embedding gather -> bf16 rows (row = b*T + t)
// ---------------------------------------------------------------------------
__global__ __launch_bounds__(256) void embed_gather_kernel(
    const int* __restrict__ tgt, const float* __restrict__ embed,
    unsigned short* __restrict__ xbf) {
  int row = blockIdx.x;                 // 0..2047
  int tid = threadIdx.x;                // 0..255
  long v = (long)tgt[row];
  const float* e = embed + v * CH;
  xbf[(long)row * CH + tid]       = f2bf(e[tid]);
  xbf[(long)row * CH + tid + 256] = f2bf(e[tid + 256]);
}

// ---------------------------------------------------------------------------
// bf16 GEMM + bias:  C[m][n] = sum_k A[m][k]*W[n][k] + bias[n]
// A: [M][K] bf16 row-major (lda==K), W: [N][ldw] bf16 row-major (acts as B^T).
// Block = 256 thr (8 waves, 4(M) x 2(N)); block tile 128x128; wave tile 32x64
// (2x4 accumulators). A K-chunks (128 rows x 64) double-buffered in LDS via
// async DMA; inner loops fully branchless so EXEC is all-ones at every WMMA.
// Requires M%128==0, N%128==0, K%64==0.  Grid = (M/128, N/128).
// ---------------------------------------------------------------------------
__global__ __launch_bounds__(256) void gemm_bf16_bias_kernel(
    const unsigned short* __restrict__ A, const unsigned short* __restrict__ W,
    const float* __restrict__ bias, float* __restrict__ C,
    int N, int K, int ldw, int ldc) {
  __shared__ __align__(16) unsigned short As[2][GMB * GKC];  // 2 x 16 KB

  const int tid   = threadIdx.x;
  const int lane  = tid & 31;
  const int wave  = tid >> 5;
  const int wm    = wave & 3;             // M group (32 rows each)
  const int wn    = wave >> 2;            // N group (64 cols each)
  const int m0    = blockIdx.x * GMB;
  const int n0    = blockIdx.y * GNB + wn * 64;
  const int lm    = lane & 15;
  const int khalf = (lane >> 4) << 3;     // 0 or 8

  // stage A chunk kc (128 rows x 64 bf16, row stride K in global) into As[buf]
  auto stage = [&](int buf, int kc) {
    const unsigned lbase = (unsigned)(uintptr_t)(&As[buf][0]);
    #pragma unroll
    for (int t = 0; t < 4; ++t) {
      int j   = tid + t * 256;            // 0..1023 b128 transfers
      int row = j >> 3;
      int cc  = j & 7;                    // 16B chunk within the 128B row
      const void* g = A + (long)(m0 + row) * K + kc * GKC + cc * 8;
      async_g2l_b128(lbase + row * (GKC * 2) + cc * 16, g);
    }
  };

  v8f acc[2][4];
  #pragma unroll
  for (int ms = 0; ms < 2; ++ms)
    #pragma unroll
    for (int ni = 0; ni < 4; ++ni) acc[ms][ni] = {};

  const unsigned short* wrow[4];
  #pragma unroll
  for (int ni = 0; ni < 4; ++ni)
    wrow[ni] = W + (long)(n0 + ni * 16 + lm) * ldw + khalf;

  stage(0, 0);
  wait_asynccnt0();
  __syncthreads();

  const int nk = K / GKC;                 // 8 for K=512
  for (int kc = 0; kc < nk; ++kc) {
    if (kc + 1 < nk) stage((kc + 1) & 1, kc + 1);
    const unsigned short* ab = As[kc & 1];
    __builtin_prefetch(wrow[0] + (kc + 1) * GKC, 0, 1);   // global_prefetch_b8
    #pragma unroll
    for (int k0 = 0; k0 < GKC; k0 += 32) {
      v16bf a0 = load_frag(ab + (wm * 32 + lm) * GKC + khalf + k0);
      v16bf a1 = load_frag(ab + (wm * 32 + 16 + lm) * GKC + khalf + k0);
      #pragma unroll
      for (int ni = 0; ni < 4; ++ni) {
        v16bf b = load_frag(wrow[ni] + kc * GKC + k0);
        acc[0][ni] = wmma_bf16(a0, b, acc[0][ni]);
        acc[1][ni] = wmma_bf16(a1, b, acc[1][ni]);
      }
    }
    wait_asynccnt0();
    __syncthreads();
  }

  #pragma unroll
  for (int ms = 0; ms < 2; ++ms) {
    const int mb = m0 + wm * 32 + ms * 16 + ((lane >> 4) << 3);
    #pragma unroll
    for (int ni = 0; ni < 4; ++ni) {
      int n = n0 + ni * 16 + lm;
      float bv = bias ? bias[n] : 0.0f;
      #pragma unroll
      for (int r = 0; r < 8; ++r)
        C[(long)(mb + r) * ldc + n] = acc[ms][ni][r] + bv;
    }
  }
}

// ---------------------------------------------------------------------------
// GRU recurrence for one layer. Single 1024-thread workgroup (32 waves).
// h [16][512] lives in LDS (f32 + bf16 mirror); h0 arrives via the TDM.
// Wave w owns h-columns c0=w*16..c0+15 and computes gh tiles for r/z/n gates
// (n0 = g*512 + c0), 16x16x(K=512) each => 48 WMMAs / wave / timestep.
// ---------------------------------------------------------------------------
__global__ __launch_bounds__(1024) void gru_layer_kernel(
    const float* __restrict__ h0,            // [16,512]
    const float* __restrict__ xg,            // [2048][1536], row = b*T + t
    const unsigned short* __restrict__ Whh,  // [1536][512] bf16
    const float* __restrict__ bhh,           // [1536]
    unsigned short* __restrict__ out_bf,     // [2048][512] bf16 or null
    float* __restrict__ out_f,               // [2048][512] f32 or null
    float* __restrict__ hT) {                // [16,512] -> hn slice in d_out
  __shared__ float hf[CB * CH];                             // 32 KB
  __shared__ __align__(16) unsigned short hb[CB * CH];      // 16 KB

  const int tid = threadIdx.x;

  // ---- h0 -> LDS via Tensor Data Mover (one descriptor, issued by wave 0) --
  if (tid < 32) {
    tdm_load_1d_f32((unsigned)(uintptr_t)(&hf[0]), h0, CB * CH);  // 32 KB
    __builtin_amdgcn_s_wait_tensorcnt(0);
  }
  __syncthreads();
  for (int i = tid; i < CB * CH; i += 1024) hb[i] = f2bf(hf[i]);
  __syncthreads();

  const int lane  = tid & 31;
  const int wave  = tid >> 5;            // 0..31
  const int lm    = lane & 15;
  const int khalf = (lane >> 4) << 3;
  const int c0    = wave * 16;
  const int c     = c0 + lm;             // owned h column
  const int mbase = (lane >> 4) << 3;    // 0 or 8

  const unsigned short* ar   = hb + lm * CH + khalf;                 // LDS A row
  const unsigned short* wr_r = Whh + (long)(c0 + lm) * CH + khalf;
  const unsigned short* wr_z = wr_r + (long)512  * CH;
  const unsigned short* wr_n = wr_r + (long)1024 * CH;
  const float brv = bhh[c0 + lm];
  const float bzv = bhh[512 + c0 + lm];
  const float bnv = bhh[1024 + c0 + lm];

  for (int t = 0; t < CT; ++t) {
    v8f accr = {}, accz = {}, accn = {};
    #pragma unroll 4
    for (int k0 = 0; k0 < CH; k0 += 32) {
      v16bf a  = load_frag(ar + k0);           // ds_load_b128 x2
      v16bf br = load_frag(wr_r + k0);
      accr = wmma_bf16(a, br, accr);
      v16bf bz = load_frag(wr_z + k0);
      accz = wmma_bf16(a, bz, accz);
      v16bf bn = load_frag(wr_n + k0);
      accn = wmma_bf16(a, bn, accn);
    }

    float hnew[8];
    unsigned short hnb[8];
    #pragma unroll
    for (int r = 0; r < 8; ++r) {
      int m = mbase + r;                       // batch row 0..15
      const float* xr = xg + (long)(m * CT + t) * C3M;
      float i_r = xr[c], i_z = xr[512 + c], i_n = xr[1024 + c];
      float rg = 1.0f / (1.0f + __expf(-(i_r + accr[r] + brv)));
      float zg = 1.0f / (1.0f + __expf(-(i_z + accz[r] + bzv)));
      float ng = tanhf(i_n + rg * (accn[r] + bnv));
      float hp = hf[m * CH + c];
      hnew[r] = (1.0f - zg) * ng + zg * hp;
      hnb[r]  = f2bf(hnew[r]);
    }
    __syncthreads();                           // all hb/hf reads done
    #pragma unroll
    for (int r = 0; r < 8; ++r) {
      int m = mbase + r;
      hf[m * CH + c] = hnew[r];
      hb[m * CH + c] = hnb[r];
    }
    if (out_bf) {                              // uniform (kernel arg)
      #pragma unroll
      for (int r = 0; r < 8; ++r)
        out_bf[(long)((mbase + r) * CT + t) * CH + c] = hnb[r];
    }
    if (out_f) {                               // uniform (kernel arg)
      #pragma unroll
      for (int r = 0; r < 8; ++r)
        out_f[(long)((mbase + r) * CT + t) * CH + c] = hnew[r];
    }
    __syncthreads();                           // new h visible for next step
  }

  for (int i = tid; i < CB * CH; i += 1024) hT[i] = hf[i];
}

// ---------------------------------------------------------------------------
// Attention: one block per (b,t) row. scores->softmax->context + residual,
// emits bf16 attn row for the projection GEMM.
// ---------------------------------------------------------------------------
__global__ __launch_bounds__(256) void attn_kernel(
    const float* __restrict__ outs,           // [2048][512]
    const float* __restrict__ enc,            // [16][256][512]
    unsigned short* __restrict__ attn_bf) {   // [2048][512]
  __shared__ float q[CH];
  __shared__ float sc[CS];
  __shared__ float red[CS];
  const int tid = threadIdx.x;
  const int row = blockIdx.x;                 // b*T + t
  const int b   = row >> 7;                   // / T

  q[tid]       = outs[(long)row * CH + tid];
  q[tid + 256] = outs[(long)row * CH + tid + 256];
  __syncthreads();

  const float* eb = enc + (long)b * CS * CH;
  const float* er = eb + (long)tid * CH;
  float acc = 0.0f;
  #pragma unroll 4
  for (int k = 0; k < CH; ++k) acc += q[k] * er[k];
  sc[tid] = acc;
  red[tid] = acc;
  __syncthreads();
  for (int s = 128; s > 0; s >>= 1) {
    if (tid < s) red[tid] = fmaxf(red[tid], red[tid + s]);
    __syncthreads();
  }
  float mx = red[0];
  __syncthreads();
  float e = __expf(sc[tid] - mx);
  red[tid] = e;
  __syncthreads();
  for (int s = 128; s > 0; s >>= 1) {
    if (tid < s) red[tid] += red[tid + s];
    __syncthreads();
  }
  float inv = 1.0f / red[0];
  __syncthreads();
  sc[tid] = e * inv;
  __syncthreads();

  #pragma unroll
  for (int half = 0; half < 2; ++half) {
    int m = tid + half * 256;
    float ctx = 0.0f;
    #pragma unroll 4
    for (int s = 0; s < CS; ++s) ctx += sc[s] * eb[(long)s * CH + m];
    attn_bf[(long)row * CH + m] = f2bf(q[m] + ctx);
  }
}

__global__ void zero_kernel(float* __restrict__ p, int n) {
  int i = blockIdx.x * blockDim.x + threadIdx.x;
  if (i < n) p[i] = 0.0f;
}

// ---------------------------------------------------------------------------
// Host launcher
// ---------------------------------------------------------------------------
extern "C" void kernel_launch(void* const* d_in, const int* in_sizes, int n_in,
                              void* d_out, int out_size, void* d_ws, size_t ws_size,
                              hipStream_t stream) {
  (void)in_sizes; (void)n_in; (void)out_size; (void)ws_size;

  const int*   tgt    = (const int*)  d_in[0];
  const float* hidden = (const float*)d_in[1];
  const float* enc    = (const float*)d_in[2];
  const float* embed  = (const float*)d_in[3];
  const float* W_ih0  = (const float*)d_in[4];
  const float* W_hh0  = (const float*)d_in[5];
  const float* b_ih0  = (const float*)d_in[6];
  const float* b_hh0  = (const float*)d_in[7];
  const float* W_ih1  = (const float*)d_in[8];
  const float* W_hh1  = (const float*)d_in[9];
  const float* b_ih1  = (const float*)d_in[10];
  const float* b_hh1  = (const float*)d_in[11];
  const float* W_out  = (const float*)d_in[12];
  const float* b_out  = (const float*)d_in[13];

  float* out = (float*)d_out;
  float* logits = out;                             // [2048][32000]
  float* hn     = out + (long)CBT * CV;            // [2][16][512]
  float* cn     = hn + 2 * CB * CH;                // zeros

  // workspace carve-up (256B aligned); total ~74 MB
  char* w = (char*)d_ws;
  size_t off = 0;
  auto carve = [&](size_t bytes) {
    void* p = w + off;
    off = (off + bytes + 255) & ~(size_t)255;
    return p;
  };
  unsigned short* xbf    = (unsigned short*)carve((size_t)CBT * CH * 2);
  unsigned short* wih0b  = (unsigned short*)carve((size_t)C3M * CH * 2);
  unsigned short* whh0b  = (unsigned short*)carve((size_t)C3M * CH * 2);
  unsigned short* wih1b  = (unsigned short*)carve((size_t)C3M * CH * 2);
  unsigned short* whh1b  = (unsigned short*)carve((size_t)C3M * CH * 2);
  unsigned short* woutb  = (unsigned short*)carve((size_t)CV  * CH * 2);
  float*          xg0    = (float*)         carve((size_t)CBT * C3M * 4);
  unsigned short* out0b  = (unsigned short*)carve((size_t)CBT * CH * 2);
  float*          xg1    = (float*)         carve((size_t)CBT * C3M * 4);
  float*          out1f  = (float*)         carve((size_t)CBT * CH * 4);
  unsigned short* attnb  = (unsigned short*)carve((size_t)CBT * CH * 2);

  // 1) weight conversions to bf16
  {
    int nw = C3M * CH;
    cvt_bf16_kernel<<<(nw + 255) / 256, 256, 0, stream>>>(W_ih0, wih0b, nw);
    cvt_bf16_kernel<<<(nw + 255) / 256, 256, 0, stream>>>(W_hh0, whh0b, nw);
    cvt_bf16_kernel<<<(nw + 255) / 256, 256, 0, stream>>>(W_ih1, wih1b, nw);
    cvt_bf16_kernel<<<(nw + 255) / 256, 256, 0, stream>>>(W_hh1, whh1b, nw);
    int nv = CV * CH;
    cvt_bf16_kernel<<<(nv + 255) / 256, 256, 0, stream>>>(W_out, woutb, nv);
  }

  // 2) embedding gather -> bf16
  embed_gather_kernel<<<CBT, 256, 0, stream>>>(tgt, embed, xbf);

  // 3) xg0 = x . W_ih0^T + b_ih0   [2048 x 1536], K=512
  gemm_bf16_bias_kernel<<<dim3(CBT / GMB, C3M / GNB), 256, 0, stream>>>(
      xbf, wih0b, b_ih0, xg0, C3M, CH, CH, C3M);

  // 4) GRU layer 0 (writes out0 bf16 + hn[0])
  gru_layer_kernel<<<1, 1024, 0, stream>>>(
      hidden, xg0, whh0b, b_hh0, out0b, (float*)nullptr, hn);

  // 5) xg1 = out0 . W_ih1^T + b_ih1
  gemm_bf16_bias_kernel<<<dim3(CBT / GMB, C3M / GNB), 256, 0, stream>>>(
      out0b, wih1b, b_ih1, xg1, C3M, CH, CH, C3M);

  // 6) GRU layer 1 (writes outputs f32 + hn[1])
  gru_layer_kernel<<<1, 1024, 0, stream>>>(
      hidden + CB * CH, xg1, whh1b, b_hh1, (unsigned short*)nullptr, out1f,
      hn + CB * CH);

  // 7) attention + residual -> bf16 attn rows
  attn_kernel<<<CBT, 256, 0, stream>>>(out1f, enc, attnb);

  // 8) logits = attn . W_out^T + b_out   [2048 x 32000], K=512
  gemm_bf16_bias_kernel<<<dim3(CBT / GMB, CV / GNB), 256, 0, stream>>>(
      attnb, woutb, b_out, logits, CV, CH, CH, CV);

  // 9) cn = zeros
  zero_kernel<<<(2 * CB * CH + 255) / 256, 256, 0, stream>>>(cn, 2 * CB * CH);
}
